// EdgeGatedGraphConv_21157008900637
// MI455X (gfx1250) — matrix operand order repaired and checked by reading, not compiled, not run on hardware
//
#include <hip/hip_runtime.h>
#include <math.h>

typedef __attribute__((ext_vector_type(16))) __bf16 v16bf;
typedef __attribute__((ext_vector_type(8)))  float  v8f;

#define WAVES 8
#define LN_EPS 1e-5f

struct Q32 { uint4 lo, hi; };

static __device__ inline unsigned short f2bf(float f) {
  unsigned u = __builtin_bit_cast(unsigned, f);
  u = (u + 0x7FFFu + ((u >> 16) & 1u)) >> 16;   // RNE to bf16
  return (unsigned short)u;
}

static __device__ inline v16bf ld_frag16(const unsigned short* p0, const unsigned short* p1) {
  Q32 q;
  q.lo = *(const uint4*)p0;   // 8 bf16 (j=0..7)
  q.hi = *(const uint4*)p1;   // 8 bf16 (j=8..15)
  return __builtin_bit_cast(v16bf, q);
}

// B fragment stored contiguously per lane in LDS: [fragIdx][lane][16 halves]
static __device__ inline v16bf bfrag(const unsigned short* sFrag, int fragIdx, int lane) {
  const unsigned short* p = sFrag + fragIdx * 512 + lane * 16;
  return ld_frag16(p, p + 8);
}

// A fragment 16x32 bf16 from one f32 row: j in [0,7] -> k = base+8*hi+j ; j in [8,15] -> k = base+16+8*hi+(j-8)
static __device__ inline v16bf afrag_row(const float* row, int base, int hi) {
  float4 f0 = *(const float4*)(row + base + 8*hi);
  float4 f1 = *(const float4*)(row + base + 8*hi + 4);
  float4 f2 = *(const float4*)(row + base + 16 + 8*hi);
  float4 f3 = *(const float4*)(row + base + 16 + 8*hi + 4);
  v16bf a;
  a[0]=(__bf16)f0.x;  a[1]=(__bf16)f0.y;  a[2]=(__bf16)f0.z;  a[3]=(__bf16)f0.w;
  a[4]=(__bf16)f1.x;  a[5]=(__bf16)f1.y;  a[6]=(__bf16)f1.z;  a[7]=(__bf16)f1.w;
  a[8]=(__bf16)f2.x;  a[9]=(__bf16)f2.y;  a[10]=(__bf16)f2.z; a[11]=(__bf16)f2.w;
  a[12]=(__bf16)f3.x; a[13]=(__bf16)f3.y; a[14]=(__bf16)f3.z; a[15]=(__bf16)f3.w;
  return a;
}

static __device__ inline v8f wmma_bf16(v16bf a, v16bf b, v8f c) {
  return __builtin_amdgcn_wmma_f32_16x16x32_bf16(false, a, false, b, (short)0, c, false, false);
}

// Pre-swizzle weight W[K][64] (row-major f32) into bf16 B fragments in LDS (pairwise: packed u32 stores).
// B-layout (32x16 bf16, dense): col = 16*n + (lane&15) ; k = 32*c + 8*(lane>>4) + j (j<8), 32*c + 16 + 8*(lane>>4) + (j-8) (j>=8)
static __device__ inline void fill_wfrags(const float* __restrict__ W, int K, int fragBase,
                                          unsigned short* sFrag) {
  unsigned* sF32 = (unsigned*)sFrag;
  const int nPairs = K * 32;
  for (int i = threadIdx.x; i < nPairs; i += blockDim.x) {
    int jp   = i & 7;            // pair index: j = 2*jp, 2*jp+1
    int lane = (i >> 3) & 31;
    int fi   = i >> 8;           // fragment index within this weight
    int n    = fi & 3;
    int c    = fi >> 2;
    int k    = 32*c + 8*(lane >> 4) + 2*jp + ((jp >> 2) << 3);
    int col  = 16*n + (lane & 15);
    float f0 = W[k * 64 + col];
    float f1 = W[(k + 1) * 64 + col];
    unsigned u0 = f2bf(f0), u1 = f2bf(f1);
    sF32[(fragBase + fi) * 256 + lane * 8 + jp] = (u1 << 16) | u0;
  }
}

static __device__ inline float sigmoidf_(float x) { return 1.0f / (1.0f + expf(-x)); }

__global__ void zero_f4_kernel(float4* __restrict__ p, int n4) {
  int i = blockIdx.x * blockDim.x + threadIdx.x;
  if (i < n4) p[i] = make_float4(0.f, 0.f, 0.f, 0.f);
}

// ---------------- Edge kernel: 2 tiles (32 edges) per wave; B frags reused for 2 WMMAs ----------------
__global__ __launch_bounds__(256) void edge_kernel(
    const float* __restrict__ node_feats, const float* __restrict__ edge_feats,
    const int* __restrict__ src, const int* __restrict__ dst,
    const float* __restrict__ Ws,  const float* __restrict__ bs,
    const float* __restrict__ We,  const float* __restrict__ be,
    const float* __restrict__ Wg,  const float* __restrict__ bg,
    const float* __restrict__ We1, const float* __restrict__ be1,
    const float* __restrict__ We2, const float* __restrict__ be2,
    const float* __restrict__ g_edge, const float* __restrict__ b_edge,
    float* __restrict__ agg, float* __restrict__ edge_out, int E)
{
  extern __shared__ unsigned short smem[];
  unsigned short* sFrag = smem;                         // 72 frags * 512 halves = 73728 B
  unsigned short* sSilu = smem + 72*512;                // WAVES * 2 tiles * 16*64 bf16 = 32768 B
  float* sBias = (float*)(smem + 72*512 + WAVES*2048);  // 6 * 64 f32

  // frag bases: Wg=0(24), We1=24(24), Ws=48(8), We=56(8), We2=64(8)
  fill_wfrags(Wg,  192, 0,  sFrag);
  fill_wfrags(We1, 192, 24, sFrag);
  fill_wfrags(Ws,  64,  48, sFrag);
  fill_wfrags(We,  64,  56, sFrag);
  fill_wfrags(We2, 64,  64, sFrag);
  for (int i = threadIdx.x; i < 64; i += blockDim.x) {
    sBias[0*64+i] = bg[i];
    sBias[1*64+i] = bs[i] + be[i];
    sBias[2*64+i] = be1[i];
    sBias[3*64+i] = be2[i];
    sBias[4*64+i] = g_edge[i];
    sBias[5*64+i] = b_edge[i];
  }
  __syncthreads();

  const int w = threadIdx.x >> 5, lane = threadIdx.x & 31;
  const int m = lane & 15, hi = lane >> 4, col16 = m;
  const int e0 = (blockIdx.x * WAVES + w) * 32;   // two 16-edge tiles
  if (e0 >= E) return;

  int eA = e0 + m;       if (eA >= E) eA = E - 1;
  int eB = e0 + 16 + m;  if (eB >= E) eB = E - 1;
  const float* prsA = node_feats + (size_t)src[eA] * 64;
  const float* prdA = node_feats + (size_t)dst[eA] * 64;
  const float* preA = edge_feats + (size_t)eA * 64;
  const float* prsB = node_feats + (size_t)src[eB] * 64;
  const float* prdB = node_feats + (size_t)dst[eB] * 64;
  const float* preB = edge_feats + (size_t)eB * 64;

  // A fragments for edge_input = [h_src | h_dst | ef], K = 192 (6 chunks of 32), both tiles
  v16bf A0[6], A1[6];
  A0[0] = afrag_row(prsA, 0, hi);  A0[1] = afrag_row(prsA, 32, hi);
  A0[2] = afrag_row(prdA, 0, hi);  A0[3] = afrag_row(prdA, 32, hi);
  A0[4] = afrag_row(preA, 0, hi);  A0[5] = afrag_row(preA, 32, hi);
  A1[0] = afrag_row(prsB, 0, hi);  A1[1] = afrag_row(prsB, 32, hi);
  A1[2] = afrag_row(prdB, 0, hi);  A1[3] = afrag_row(prdB, 32, hi);
  A1[4] = afrag_row(preB, 0, hi);  A1[5] = afrag_row(preB, 32, hi);

  int drA[8], drB[8];
  #pragma unroll
  for (int r = 0; r < 8; ++r) {
    int er = e0 + r + 8*hi;      if (er >= E) er = E - 1;
    drA[r] = dst[er];
    er = e0 + 16 + r + 8*hi;     if (er >= E) er = E - 1;
    drB[r] = dst[er];
  }

  unsigned short* ssw0 = sSilu + w * 2048;
  unsigned short* ssw1 = ssw0 + 1024;

  #pragma unroll
  for (int n = 0; n < 4; ++n) {
    v8f ag0 = {}, ag1 = {}, au0 = {}, au1 = {}, al0 = {}, al1 = {};
    #pragma unroll
    for (int c = 0; c < 6; ++c) {
      v16bf b = bfrag(sFrag, 0 + c*4 + n, lane);
      ag0 = wmma_bf16(A0[c], b, ag0);
      ag1 = wmma_bf16(A1[c], b, ag1);
    }
    #pragma unroll
    for (int c = 0; c < 6; ++c) {
      v16bf b = bfrag(sFrag, 24 + c*4 + n, lane);
      au0 = wmma_bf16(A0[c], b, au0);
      au1 = wmma_bf16(A1[c], b, au1);
    }
    {
      v16bf b = bfrag(sFrag, 48 + 0*4 + n, lane);
      al0 = wmma_bf16(A0[0], b, al0);  al1 = wmma_bf16(A1[0], b, al1);
      b = bfrag(sFrag, 48 + 1*4 + n, lane);
      al0 = wmma_bf16(A0[1], b, al0);  al1 = wmma_bf16(A1[1], b, al1);
      b = bfrag(sFrag, 56 + 0*4 + n, lane);
      al0 = wmma_bf16(A0[4], b, al0);  al1 = wmma_bf16(A1[4], b, al1);
      b = bfrag(sFrag, 56 + 1*4 + n, lane);
      al0 = wmma_bf16(A0[5], b, al0);  al1 = wmma_bf16(A1[5], b, al1);
    }

    const int col = 16*n + col16;
    const float bgc = sBias[0*64+col], bmc = sBias[1*64+col], b1c = sBias[2*64+col];
    #pragma unroll
    for (int r = 0; r < 8; ++r) {
      int erA = e0 + r + 8*hi;
      float gate = sigmoidf_(ag0[r] + bgc);
      float msg  = gate * (al0[r] + bmc);
      if (erA < E) atomicAdd(&agg[(size_t)drA[r]*64 + col], msg);
      float u = au0[r] + b1c;
      ssw0[(r + 8*hi)*64 + col] = f2bf(u * sigmoidf_(u));

      int erB = e0 + 16 + r + 8*hi;
      gate = sigmoidf_(ag1[r] + bgc);
      msg  = gate * (al1[r] + bmc);
      if (erB < E) atomicAdd(&agg[(size_t)drB[r]*64 + col], msg);
      u = au1[r] + b1c;
      ssw1[(r + 8*hi)*64 + col] = f2bf(u * sigmoidf_(u));
    }
  }

  asm volatile("s_wait_dscnt 0" ::: "memory");   // wave-local LDS RAW

  // Rebuild SiLU tiles as A operands (K = 64, 2 chunks each)
  v16bf C00 = ld_frag16(ssw0 + m*64 + 8*hi,      ssw0 + m*64 + 16 + 8*hi);
  v16bf C01 = ld_frag16(ssw0 + m*64 + 32 + 8*hi, ssw0 + m*64 + 48 + 8*hi);
  v16bf C10 = ld_frag16(ssw1 + m*64 + 8*hi,      ssw1 + m*64 + 16 + 8*hi);
  v16bf C11 = ld_frag16(ssw1 + m*64 + 32 + 8*hi, ssw1 + m*64 + 48 + 8*hi);

  float x[2][4][8];
  #pragma unroll
  for (int n = 0; n < 4; ++n) {
    v8f acc0 = {}, acc1 = {};
    v16bf b = bfrag(sFrag, 64 + 0*4 + n, lane);
    acc0 = wmma_bf16(C00, b, acc0);
    acc1 = wmma_bf16(C10, b, acc1);
    b = bfrag(sFrag, 64 + 1*4 + n, lane);
    acc0 = wmma_bf16(C01, b, acc0);
    acc1 = wmma_bf16(C11, b, acc1);
    const int col = 16*n + col16;
    const float b2c = sBias[3*64+col];
    #pragma unroll
    for (int r = 0; r < 8; ++r) {
      int er = e0 + r + 8*hi; if (er >= E) er = E - 1;
      x[0][n][r] = acc0[r] + b2c + edge_feats[(size_t)er*64 + col];
      er = e0 + 16 + r + 8*hi; if (er >= E) er = E - 1;
      x[1][n][r] = acc1[r] + b2c + edge_feats[(size_t)er*64 + col];
    }
  }

  float gc[4], bc[4];
  #pragma unroll
  for (int n = 0; n < 4; ++n) { gc[n] = sBias[4*64 + 16*n + col16]; bc[n] = sBias[5*64 + 16*n + col16]; }

  #pragma unroll
  for (int t = 0; t < 2; ++t) {
    #pragma unroll
    for (int r = 0; r < 8; ++r) {
      float s  = x[t][0][r] + x[t][1][r] + x[t][2][r] + x[t][3][r];
      float ss = x[t][0][r]*x[t][0][r] + x[t][1][r]*x[t][1][r]
               + x[t][2][r]*x[t][2][r] + x[t][3][r]*x[t][3][r];
      #pragma unroll
      for (int msk = 1; msk < 16; msk <<= 1) {
        s  += __shfl_xor(s,  msk, 32);
        ss += __shfl_xor(ss, msk, 32);
      }
      float mu  = s * (1.0f/64.0f);
      float var = ss * (1.0f/64.0f) - mu*mu;
      float inv = rsqrtf(var + LN_EPS);
      int er = e0 + 16*t + r + 8*hi;
      if (er < E) {
        float* po = edge_out + (size_t)er*64 + col16;
        #pragma unroll
        for (int n = 0; n < 4; ++n) po[16*n] = (x[t][n][r] - mu) * inv * gc[n] + bc[n];
      }
    }
  }
}

// ---------------- Node kernel: [node_feats | agg] MLP + residual LayerNorm ----------------
__global__ __launch_bounds__(256) void node_kernel(
    const float* __restrict__ node_feats, const float* __restrict__ agg,
    const float* __restrict__ Wn1, const float* __restrict__ bn1,
    const float* __restrict__ Wn2, const float* __restrict__ bn2,
    const float* __restrict__ g_node, const float* __restrict__ b_node,
    float* __restrict__ node_out, int N)
{
  __shared__ unsigned short sFrag[24*512];       // Wn1: 16 frags, Wn2: 8 frags
  __shared__ unsigned short sSilu[WAVES*1024];
  __shared__ float sBias[4*64];

  fill_wfrags(Wn1, 128, 0,  sFrag);
  fill_wfrags(Wn2, 64,  16, sFrag);
  for (int i = threadIdx.x; i < 64; i += blockDim.x) {
    sBias[0*64+i] = bn1[i];
    sBias[1*64+i] = bn2[i];
    sBias[2*64+i] = g_node[i];
    sBias[3*64+i] = b_node[i];
  }
  __syncthreads();

  const int w = threadIdx.x >> 5, lane = threadIdx.x & 31;
  const int m = lane & 15, hi = lane >> 4, col16 = m;
  const int v0 = (blockIdx.x * WAVES + w) * 16;
  if (v0 >= N) return;

  int v = v0 + m; if (v >= N) v = N - 1;
  const float* pn = node_feats + (size_t)v * 64;
  const float* pa = agg        + (size_t)v * 64;

  v16bf A[4];
  A[0] = afrag_row(pn, 0, hi);  A[1] = afrag_row(pn, 32, hi);
  A[2] = afrag_row(pa, 0, hi);  A[3] = afrag_row(pa, 32, hi);

  unsigned short* ssw = sSilu + w * 1024;

  #pragma unroll
  for (int n = 0; n < 4; ++n) {
    v8f acc = {};
    #pragma unroll
    for (int c = 0; c < 4; ++c) acc = wmma_bf16(A[c], bfrag(sFrag, c*4 + n, lane), acc);
    const int col = 16*n + col16;
    const float b1c = sBias[0*64+col];
    #pragma unroll
    for (int r = 0; r < 8; ++r) {
      float u = acc[r] + b1c;
      ssw[(r + 8*hi)*64 + col] = f2bf(u * sigmoidf_(u));
    }
  }

  asm volatile("s_wait_dscnt 0" ::: "memory");

  v16bf C0 = ld_frag16(ssw + m*64 + 8*hi,      ssw + m*64 + 16 + 8*hi);
  v16bf C1 = ld_frag16(ssw + m*64 + 32 + 8*hi, ssw + m*64 + 48 + 8*hi);

  float x[4][8];
  #pragma unroll
  for (int n = 0; n < 4; ++n) {
    v8f acc = {};
    acc = wmma_bf16(C0, bfrag(sFrag, 16 + 0*4 + n, lane), acc);
    acc = wmma_bf16(C1, bfrag(sFrag, 16 + 1*4 + n, lane), acc);
    const int col = 16*n + col16;
    const float b2c = sBias[1*64+col];
    #pragma unroll
    for (int r = 0; r < 8; ++r) {
      int vr = v0 + r + 8*hi; if (vr >= N) vr = N - 1;
      x[n][r] = acc[r] + b2c + node_feats[(size_t)vr*64 + col];  // residual
    }
  }

  float gc[4], bc[4];
  #pragma unroll
  for (int n = 0; n < 4; ++n) { gc[n] = sBias[2*64 + 16*n + col16]; bc[n] = sBias[3*64 + 16*n + col16]; }

  #pragma unroll
  for (int r = 0; r < 8; ++r) {
    float s  = x[0][r] + x[1][r] + x[2][r] + x[3][r];
    float ss = x[0][r]*x[0][r] + x[1][r]*x[1][r] + x[2][r]*x[2][r] + x[3][r]*x[3][r];
    #pragma unroll
    for (int msk = 1; msk < 16; msk <<= 1) {
      s  += __shfl_xor(s,  msk, 32);
      ss += __shfl_xor(ss, msk, 32);
    }
    float mu  = s * (1.0f/64.0f);
    float var = ss * (1.0f/64.0f) - mu*mu;
    float inv = rsqrtf(var + LN_EPS);
    int vr = v0 + r + 8*hi;
    if (vr < N) {
      float* po = node_out + (size_t)vr*64 + col16;
      #pragma unroll
      for (int n = 0; n < 4; ++n) po[16*n] = (x[n][r] - mu) * inv * gc[n] + bc[n];
    }
  }
}

extern "C" void kernel_launch(void* const* d_in, const int* in_sizes, int n_in,
                              void* d_out, int out_size, void* d_ws, size_t ws_size,
                              hipStream_t stream) {
  (void)n_in; (void)out_size; (void)ws_size;
  const float* node_feats = (const float*)d_in[0];
  const float* edge_feats = (const float*)d_in[1];
  const int*   src        = (const int*)d_in[2];
  const int*   dst        = (const int*)d_in[3];
  const float* Ws  = (const float*)d_in[4];
  const float* bs  = (const float*)d_in[5];
  const float* We  = (const float*)d_in[6];
  const float* be  = (const float*)d_in[7];
  const float* Wg  = (const float*)d_in[8];
  const float* bg  = (const float*)d_in[9];
  const float* Wn1 = (const float*)d_in[10];
  const float* bn1 = (const float*)d_in[11];
  const float* Wn2 = (const float*)d_in[12];
  const float* bn2 = (const float*)d_in[13];
  const float* We1 = (const float*)d_in[14];
  const float* be1 = (const float*)d_in[15];
  const float* We2 = (const float*)d_in[16];
  const float* be2 = (const float*)d_in[17];
  const float* g_node = (const float*)d_in[18];
  const float* b_node = (const float*)d_in[19];
  const float* g_edge = (const float*)d_in[20];
  const float* b_edge = (const float*)d_in[21];

  const int N = in_sizes[0] / 64;
  const int E = in_sizes[2];

  float* node_out = (float*)d_out;
  float* edge_out = node_out + (size_t)N * 64;
  float* agg      = (float*)d_ws;            // N*64 f32 scratch for segment_sum

  // 1) zero agg
  int n4 = N * 16;                            // N*64/4 float4s
  zero_f4_kernel<<<(n4 + 255) / 256, 256, 0, stream>>>((float4*)agg, n4);

  // 2) edge kernel (dynamic LDS: 72 B-frags + per-wave 2-tile SiLU staging + biases)
  size_t smemE = (size_t)(72*512 + WAVES*2048) * sizeof(unsigned short) + 6*64*sizeof(float);
  int eb = (E + WAVES*32 - 1) / (WAVES*32);
  edge_kernel<<<eb, 256, smemE, stream>>>(node_feats, edge_feats, src, dst,
                                          Ws, bs, We, be, Wg, bg, We1, be1, We2, be2,
                                          g_edge, b_edge, agg, edge_out, E);

  // 3) node kernel
  int nb = (N + WAVES*16 - 1) / (WAVES*16);
  node_kernel<<<nb, 256, 0, stream>>>(node_feats, agg, Wn1, bn1, Wn2, bn2,
                                      g_node, b_node, node_out, N);
}
